// EstraNet_1443109012284
// MI455X (gfx1250) — matrix operand order repaired and checked by reading, not compiled, not run on hardware
//
#include <hip/hip_runtime.h>

// ============================================================================
// EstraNet forward, MI455X (gfx1250, wave32, WMMA + async-to-LDS).
//
// The reference's FAVOR+ branch is dead code: out = x @ M with
//   M[d',d]  = sum_{h,c} W_v[d',h,c] * norma[h] * W_o[h,c,d]
//   norma[h] = || s_p[h] * sum_d W_p[d,h,c] * beta_p[d] ||_c
// Main GEMM is [32768,512]x[512,512]: 17.2 GFLOP vs 128 MB traffic
// (~5.5us at 23.3 TB/s). v_wmma_f32_16x16x32_f16 with split-f16 (hi+lo)
// x and M (3 WMMAs/tile, f32 accumulate) keeps it memory-bound at ~f32
// accuracy. B tiles (pre-converted f16) are staged with
// GLOBAL_LOAD_ASYNC_TO_LDS_B128 (ASYNCcnt) overlapping the VALU f32->f16
// split of the A tile.
// ============================================================================

typedef __attribute__((ext_vector_type(16))) _Float16 v16h;
typedef __attribute__((ext_vector_type(8)))  _Float16 v8h;
typedef __attribute__((ext_vector_type(8)))  float    v8f;

#define D_DIM 512
#define NROWS 32768   // B*N = 8*4096
#define BM 128
#define BN 128
#define BK 64

// ---------------------------------------------------------------------------
// Kernel 1: norma[h] (8 floats). One block of 512 threads, t = h*64 + c.
// ---------------------------------------------------------------------------
__global__ __launch_bounds__(512) void norma_kernel(
    const float* __restrict__ Wp,      // [D, NC, DC] = [512, 8, 64]
    const float* __restrict__ beta_p,  // [512]
    const float* __restrict__ s_p,     // [8]
    float* __restrict__ norma)         // [8]
{
  __shared__ float sh[512];
  const int t = threadIdx.x;
  float acc = 0.f;
  for (int d = 0; d < D_DIM; ++d)
    acc = fmaf(Wp[d * 512 + t], beta_p[d], acc);   // coalesced over t
  const float val = s_p[t >> 6] * acc;
  sh[t] = val * val;
  __syncthreads();
  if ((t & 63) == 0) {
    float s = 0.f;
    for (int c = 0; c < 64; ++c) s += sh[t + c];
    norma[t >> 6] = sqrtf(s);
  }
}

// ---------------------------------------------------------------------------
// Kernel 2: fused matrix, stored TRANSPOSED (n-major, k-minor) and split into
// f16 hi/lo:  Mt[n][k] = sum_hc W_v[k,hc] * norma[hc/64] * W_o[hc,n].
// 134 MFLOP total — negligible; plain VALU.
// ---------------------------------------------------------------------------
__global__ __launch_bounds__(256) void build_M(
    const float* __restrict__ Wv,      // [512, 512] (d', hc)
    const float* __restrict__ Wo,      // [512, 512] (hc, d)
    const float* __restrict__ norma,   // [8]
    _Float16* __restrict__ MtHi,       // [512, 512] n-major
    _Float16* __restrict__ MtLo)
{
  const int idx = blockIdx.x * 256 + threadIdx.x;
  const int k = idx >> 9;              // input dim d' (uniform per block)
  const int n = idx & 511;             // output dim d (coalesced per lane)
  float g[8];
  #pragma unroll
  for (int i = 0; i < 8; ++i) g[i] = norma[i];
  const float* wvRow = Wv + (size_t)k * 512;   // scalar-broadcast reads
  float acc = 0.f;
  for (int hc = 0; hc < 512; ++hc)
    acc = fmaf(wvRow[hc] * g[hc >> 6], Wo[(size_t)hc * 512 + n], acc);
  const _Float16 h = (_Float16)acc;
  MtHi[(size_t)n * 512 + k] = h;
  MtLo[(size_t)n * 512 + k] = (_Float16)(acc - (float)h);
}

// ---------------------------------------------------------------------------
// Async global->LDS copy of 16 bytes (GLOBAL_LOAD_ASYNC_TO_LDS_B128).
// dst LDS byte offset comes from the low 32 bits of the generic pointer
// (flat LDS addressing, ISA 10.2); the async unit adds
// the wave's LDS base. Tracked by ASYNCcnt.
// ---------------------------------------------------------------------------
__device__ __forceinline__ void async_copy_b128(void* lds_dst, const void* gsrc) {
  const unsigned lds_off = (unsigned)(unsigned long long)lds_dst;
  asm volatile("global_load_async_to_lds_b128 %0, %1, off"
               :
               : "v"(lds_off), "v"(gsrc)
               : "memory");
}
__device__ __forceinline__ void async_wait0() {
  asm volatile("s_wait_asynccnt 0x0" ::: "memory");
}

// ---------------------------------------------------------------------------
// Kernel 3: out[32768,512] = x[32768,512] @ M[512,512], split-f16 WMMA.
// Block = 256 threads = 8 waves; 128x128 C tile; BK=64 (2 WMMA k-steps).
// Wave (wr,wc) in 4x2 grid owns a 32x64 patch = 2x4 WMMA 16x16 tiles.
// LDS fragment loads use contiguous k order (k = 16*(lane>>4) + s) for A(16x32)/B(32x16):
//   A lane (m=lane&15, kb=(lane>>4)*16): halves [kb..kb+15] of row m
//   B lane (n=lane&15, kb=(lane>>4)*16): halves [kb..kb+15] of column n
// B is staged n-major so the 16-half lane chunk is one contiguous 32B load.
// ---------------------------------------------------------------------------
__device__ __forceinline__ v8f wmma_f16(v16h a, v16h b, v8f c) {
  return __builtin_amdgcn_wmma_f32_16x16x32_f16(
      /*neg_a=*/false, a, /*neg_b=*/false, b,
      /*c_mod=*/(short)0, c, /*reuse_a=*/false, /*reuse_b=*/false);
}

__global__ __launch_bounds__(256) void gemm_kernel(
    const float* __restrict__ X,         // [32768, 512]
    const _Float16* __restrict__ MtHi,   // [512, 512] n-major
    const _Float16* __restrict__ MtLo,
    float* __restrict__ Out)             // [32768, 512]
{
  __shared__ _Float16 AsHi[BM * BK];   // 16 KB
  __shared__ _Float16 AsLo[BM * BK];   // 16 KB
  __shared__ _Float16 BsHi[BN * BK];   // 16 KB (n-major)
  __shared__ _Float16 BsLo[BN * BK];   // 16 KB

  const int tid   = threadIdx.x;
  const int lane  = tid & 31;
  const int wave  = tid >> 5;
  const int wr    = wave >> 1;         // 0..3: wave row
  const int wc    = wave & 1;          // 0..1: wave col
  const int lm    = lane & 15;
  const int lhalf = lane >> 4;
  const size_t rowBase = (size_t)blockIdx.x * BM;
  const int    colBase = blockIdx.y * BN;

  const v8f vzero = {0.f, 0.f, 0.f, 0.f, 0.f, 0.f, 0.f, 0.f};
  v8f acc[2][4];
  #pragma unroll
  for (int mr = 0; mr < 2; ++mr)
    #pragma unroll
    for (int nc = 0; nc < 4; ++nc)
      acc[mr][nc] = vzero;

  for (int kt = 0; kt < D_DIM; kt += BK) {
    // ---- B tiles: async global->LDS (f16, no conversion needed).
    // Issued first so the async engine overlaps the A-tile VALU split below.
    #pragma unroll
    for (int it = 0; it < 4; ++it) {
      const int linear = it * 2048 + tid * 8;
      const int r  = linear >> 6;
      const int cL = linear & 63;
      const size_t g = (size_t)(colBase + r) * D_DIM + kt + cL;
      async_copy_b128(&BsHi[r * BK + cL], MtHi + g);
      async_copy_b128(&BsLo[r * BK + cL], MtLo + g);
    }

    // ---- A tile: 128x64 f32 -> hi/lo f16 in LDS (needs VALU conversion) ----
    #pragma unroll
    for (int it = 0; it < 8; ++it) {
      const int linear = it * 1024 + tid * 4;
      const int r  = linear >> 6;
      const int cL = linear & 63;
      const float4 v = *(const float4*)(X + (rowBase + r) * D_DIM + kt + cL);
      const int o = r * BK + cL;
      const _Float16 h0 = (_Float16)v.x; AsHi[o+0]=h0; AsLo[o+0]=(_Float16)(v.x-(float)h0);
      const _Float16 h1 = (_Float16)v.y; AsHi[o+1]=h1; AsLo[o+1]=(_Float16)(v.y-(float)h1);
      const _Float16 h2 = (_Float16)v.z; AsHi[o+2]=h2; AsLo[o+2]=(_Float16)(v.z-(float)h2);
      const _Float16 h3 = (_Float16)v.w; AsHi[o+3]=h3; AsLo[o+3]=(_Float16)(v.w-(float)h3);
    }

    async_wait0();        // ASYNCcnt == 0: B tiles landed in LDS
    __syncthreads();

    if (kt + BK < D_DIM) {
      // gfx1250 global_prefetch_b8: pull next A tile toward the caches
      __builtin_prefetch(X + (rowBase + (tid >> 1)) * D_DIM + (kt + BK) + (tid & 1) * 32, 0, 3);
    }

    #pragma unroll
    for (int ks = 0; ks < 2; ++ks) {
      // A fragments (hi/lo) for both row tiles of this wave
      v16h aHi[2], aLo[2];
      #pragma unroll
      for (int mr = 0; mr < 2; ++mr) {
        const int row  = wr * 32 + mr * 16 + lm;
        const int base = row * BK + ks * 32 + lhalf * 16;
        const v8h p0 = *(const v8h*)&AsHi[base];
        const v8h p1 = *(const v8h*)&AsHi[base + 8];
        aHi[mr] = __builtin_shufflevector(p0, p1, 0,1,2,3,4,5,6,7,8,9,10,11,12,13,14,15);
        const v8h q0 = *(const v8h*)&AsLo[base];
        const v8h q1 = *(const v8h*)&AsLo[base + 8];
        aLo[mr] = __builtin_shufflevector(q0, q1, 0,1,2,3,4,5,6,7,8,9,10,11,12,13,14,15);
      }
      #pragma unroll
      for (int nc = 0; nc < 4; ++nc) {
        const int n    = wc * 64 + nc * 16 + lm;
        const int base = n * BK + ks * 32 + lhalf * 16;
        const v16h bHi = *(const v16h*)&BsHi[base];
        const v16h bLo = *(const v16h*)&BsLo[base];
        #pragma unroll
        for (int mr = 0; mr < 2; ++mr) {
          acc[mr][nc] = wmma_f16(aHi[mr], bHi, acc[mr][nc]);  // hi*hi
          acc[mr][nc] = wmma_f16(aLo[mr], bHi, acc[mr][nc]);  // lo*hi
          acc[mr][nc] = wmma_f16(aHi[mr], bLo, acc[mr][nc]);  // hi*lo
        }
      }
    }
    __syncthreads();
  }

  // ---- store C (f32 16x16 layout: VGPR r -> M = r + (lane>>4)*8, N = lane&15)
  #pragma unroll
  for (int mr = 0; mr < 2; ++mr) {
    #pragma unroll
    for (int nc = 0; nc < 4; ++nc) {
      const int col = colBase + wc * 64 + nc * 16 + lm;
      #pragma unroll
      for (int r = 0; r < 8; ++r) {
        const size_t row = rowBase + wr * 32 + mr * 16 + lhalf * 8 + r;
        Out[row * D_DIM + col] = acc[mr][nc][r];
      }
    }
  }
}

// ---------------------------------------------------------------------------
// Launch. Inputs (setup_inputs order):
//   0:x 1:W_v 2:s_p 3:c_p 4:W_p 5:W_A 6:W_o 7:beta_p 8:beta_i_p  (all f32)
// Workspace: [0,32)  norma (8 f32)
//            [256, 256+512KB)        MtHi (512*512 f16)
//            [256+512KB, 256+1MB)    MtLo
// ---------------------------------------------------------------------------
extern "C" void kernel_launch(void* const* d_in, const int* in_sizes, int n_in,
                              void* d_out, int out_size, void* d_ws, size_t ws_size,
                              hipStream_t stream) {
  (void)in_sizes; (void)n_in; (void)out_size; (void)ws_size;

  const float* x   = (const float*)d_in[0];
  const float* Wv  = (const float*)d_in[1];
  const float* sp  = (const float*)d_in[2];
  const float* Wp  = (const float*)d_in[4];
  const float* Wo  = (const float*)d_in[6];
  const float* bp  = (const float*)d_in[7];
  float* out = (float*)d_out;

  float*    norma = (float*)d_ws;
  _Float16* MtHi  = (_Float16*)((char*)d_ws + 256);
  _Float16* MtLo  = (_Float16*)((char*)d_ws + 256 + (size_t)D_DIM * D_DIM * sizeof(_Float16));

  norma_kernel<<<1, 512, 0, stream>>>(Wp, bp, sp, norma);
  build_M<<<(D_DIM * D_DIM) / 256, 256, 0, stream>>>(Wv, Wo, norma, MtHi, MtLo);
  dim3 grid(NROWS / BM, D_DIM / BN);
  gemm_kernel<<<grid, 256, 0, stream>>>(x, MtHi, MtLo, out);
}
